// MultiHeadAttention_83245056131661
// MI455X (gfx1250) — matrix operand order repaired and checked by reading, not compiled, not run on hardware
//
#include <hip/hip_runtime.h>
#include <hip/hip_bf16.h>

typedef __bf16 bf16;
typedef __attribute__((ext_vector_type(8)))  __bf16 v8bf;
typedef __attribute__((ext_vector_type(16))) __bf16 v16bf;
typedef __attribute__((ext_vector_type(8)))  float  v8f;

#define BDIM   2
#define SDIM   4096
#define HEADS  12
#define DK     64
#define DM     768
#define MROWS  (BDIM * SDIM)          // 8192

__device__ __forceinline__ v16bf ld16(const bf16* p0, const bf16* p1) {
    v8bf lo = *(const v8bf*)p0;
    v8bf hi = *(const v8bf*)p1;
    return __builtin_shufflevector(lo, hi, 0,1,2,3,4,5,6,7,8,9,10,11,12,13,14,15);
}

__device__ __forceinline__ v8f wmma_bf16(v16bf a, v16bf b, v8f c) {
    return __builtin_amdgcn_wmma_f32_16x16x32_bf16(false, a, false, b, (short)0, c, false, false);
}

// gfx1250 async global->LDS copy (16B per lane), tracked by ASYNCcnt
__device__ __forceinline__ void async_copy_b128(const bf16* gptr, bf16* lptr) {
    unsigned lds_off = (unsigned)(size_t)lptr;          // low 32 bits = LDS offset
    unsigned long long ga = (unsigned long long)(size_t)gptr;
    asm volatile("global_load_async_to_lds_b128 %0, %1, off"
                 :: "v"(lds_off), "v"(ga) : "memory");
}
__device__ __forceinline__ void wait_async0() {
    asm volatile("s_wait_asynccnt 0x0" ::: "memory");
}

// ---------------------------------------------------------------- fp32 -> bf16
__global__ void cvt_kernel(const float* __restrict__ in, bf16* __restrict__ out, int n) {
    int i = blockIdx.x * blockDim.x + threadIdx.x;
    int stride = gridDim.x * blockDim.x;
    for (; i < n; i += stride) out[i] = (bf16)in[i];
}

// ---------------------------------------------------------------- projection GEMM
// C[m,n] = sum_k A[m,k] * W[n,k] + bias[n]   (A: MROWS x 768 bf16, W: 768x768 bf16)
// MODE 0: store bf16 head-split   out[(b*H+h)*S*64 + s*64 + d]
// MODE 2: store bf16 transposed   out[((b*H+h)*64 + d)*S + s]   (for V)
// MODE 3: store fp32 row-major    out[m*768 + n]                (final proj)
template <int MODE>
__global__ __launch_bounds__(256) void proj_kernel(const bf16* __restrict__ A,
                                                   const bf16* __restrict__ W,
                                                   const float* __restrict__ bias,
                                                   void* __restrict__ outp) {
    const int lane   = threadIdx.x & 31;
    const int wid    = blockIdx.x * (blockDim.x >> 5) + (threadIdx.x >> 5);
    const int NSTRIP = DM / 64;                       // 12 strips of 64 cols
    const int m_tile = wid / NSTRIP;
    const int n_base = (wid % NSTRIP) * 64;
    const int l15    = lane & 15;
    const int hiHalf = (lane >= 16) ? 1 : 0;

    v8f acc[4] = {};
    const bf16* Arow = A + (size_t)(m_tile * 16 + l15) * DM;

    for (int kb = 0; kb < DM / 32; ++kb) {
        const int ka = kb * 32 + (hiHalf ? 8 : 0);
        v16bf av = ld16(Arow + ka, Arow + ka + 16);
#pragma unroll
        for (int ns = 0; ns < 4; ++ns) {
            const int n = n_base + ns * 16 + l15;
            const bf16* Wp = W + (size_t)n * DM + kb * 32 + (hiHalf ? 8 : 0);
            v16bf bv = ld16(Wp, Wp + 16);
            acc[ns] = wmma_bf16(av, bv, acc[ns]);
        }
    }

#pragma unroll
    for (int ns = 0; ns < 4; ++ns) {
        const int n = n_base + ns * 16 + l15;
        const float bsv = bias[n];
#pragma unroll
        for (int j = 0; j < 8; ++j) {
            const int m = m_tile * 16 + j + (hiHalf ? 8 : 0);
            const float val = acc[ns][j] + bsv;
            if (MODE == 3) {
                ((float*)outp)[(size_t)m * DM + n] = val;
            } else {
                const int bb = m >> 12, s = m & (SDIM - 1);
                const int h = n >> 6, d = n & 63;
                if (MODE == 2)
                    ((bf16*)outp)[((size_t)(bb * HEADS + h) * DK + d) * SDIM + s] = (bf16)val;
                else
                    ((bf16*)outp)[((size_t)(bb * HEADS + h) * SDIM + s) * DK + d] = (bf16)val;
            }
        }
    }
}

// ---------------------------------------------------------------- flash attention
// 4 waves / block, 64 queries / block, keys in chunks of 32 staged in LDS via
// async global->LDS copies shared by all 4 waves (4x K/V reuse).
// Q,K: [B*H, S, 64] bf16 row-major.  Vt: [B*H, 64, S] bf16.  attn: [B*S, 768] bf16.
__global__ __launch_bounds__(128) void attn_kernel(const bf16* __restrict__ Q,
                                                   const bf16* __restrict__ K,
                                                   const bf16* __restrict__ Vt,
                                                   bf16* __restrict__ attn) {
    __shared__ bf16 Kl[32][64];          // 4 KB: K chunk, row-major [key][d]
    __shared__ bf16 Vl[64][32];          // 4 KB: V chunk, [d][key]
    __shared__ bf16 Pl[4][16][32];       // 4 KB: per-wave P transpose buffers

    const int tid    = threadIdx.x;
    const int lane   = tid & 31;
    const int wv     = tid >> 5;                     // wave 0..3
    const int bh     = blockIdx.x >> 6;              // / (S/64)
    const int q0     = (blockIdx.x & 63) * 64 + wv * 16;
    const int l15    = lane & 15;
    const int hiHalf = (lane >= 16) ? 1 : 0;

    // Q A-operands: two K=32 chunks covering d=0..63
    const bf16* Qrow = Q + ((size_t)bh * SDIM + q0 + l15) * DK;
    v16bf qa[2];
#pragma unroll
    for (int kc = 0; kc < 2; ++kc) {
        const int dbase = kc * 32 + (hiHalf ? 8 : 0);
        qa[kc] = ld16(Qrow + dbase, Qrow + dbase + 16);
    }

    v16bf ones;
#pragma unroll
    for (int i = 0; i < 16; ++i) ones[i] = (bf16)1.0f;

    float mrun[8], lrun[8];
#pragma unroll
    for (int j = 0; j < 8; ++j) { mrun[j] = -1e30f; lrun[j] = 0.0f; }
    v8f O[4] = {};

    const float scale = 0.125f;                      // 1/sqrt(64)
    for (int kt = 0; kt < SDIM / 32; ++kt) {
        const int k0 = kt * 32;

        __syncthreads();                             // prior LDS consumers done
        // stage K chunk (32 x 64 bf16) and V chunk (64 x 32 bf16) via async DMA
#pragma unroll
        for (int i = 0; i < 2; ++i) {
            const int c = tid + i * 128;             // 256 x 16B chunks each
            const int kr = c >> 3, kcol = c & 7;     // K: 32 rows x 8 chunks
            async_copy_b128(K + ((size_t)bh * SDIM + k0 + kr) * DK + kcol * 8,
                            &Kl[kr][kcol * 8]);
            const int vr = c >> 2, vcol = c & 3;     // V: 64 rows x 4 chunks
            async_copy_b128(Vt + ((size_t)bh * DK + vr) * SDIM + k0 + vcol * 8,
                            &Vl[vr][vcol * 8]);
        }
        wait_async0();
        __syncthreads();                             // chunk visible to all waves

        // scores = Q . K^T  (16 q x 32 keys), B-operands from LDS
        v8f s0 = {}, s1 = {};
#pragma unroll
        for (int kc = 0; kc < 2; ++kc) {
            const int dbase = kc * 32 + (hiHalf ? 8 : 0);
            const bf16* kp0 = &Kl[l15][dbase];
            s0 = wmma_bf16(qa[kc], ld16(kp0, kp0 + 16), s0);
            const bf16* kp1 = &Kl[16 + l15][dbase];
            s1 = wmma_bf16(qa[kc], ld16(kp1, kp1 + 16), s1);
        }

        // online softmax: row max via cross-lane reduce, p = exp(s - m)
        float p0[8], p1[8], corr[8];
#pragma unroll
        for (int j = 0; j < 8; ++j) {
            const float a0 = s0[j] * scale, a1 = s1[j] * scale;
            float t = fmaxf(a0, a1);
            t = fmaxf(t, __shfl_xor(t, 1, 32));
            t = fmaxf(t, __shfl_xor(t, 2, 32));
            t = fmaxf(t, __shfl_xor(t, 4, 32));
            t = fmaxf(t, __shfl_xor(t, 8, 32));      // row max across 16-lane half
            const float nm = fmaxf(mrun[j], t);
            corr[j] = __expf(mrun[j] - nm);
            p0[j] = __expf(a0 - nm);
            p1[j] = __expf(a1 - nm);
            mrun[j] = nm;
        }

        // transpose P (16 q x 32 k) through LDS into A-operand layout
#pragma unroll
        for (int j = 0; j < 8; ++j) {
            const int r = j + (hiHalf ? 8 : 0);
            Pl[wv][r][l15]      = (bf16)p0[j];
            Pl[wv][r][16 + l15] = (bf16)p1[j];
        }
        const int ko = hiHalf ? 8 : 0;
        v16bf pa = ld16(&Pl[wv][l15][ko], &Pl[wv][l15][ko + 16]);

        // row sums via P x ones (lane-replicated in C layout) - no shuffles
        v8f rs = {};
        rs = wmma_bf16(pa, ones, rs);
#pragma unroll
        for (int j = 0; j < 8; ++j) lrun[j] = lrun[j] * corr[j] + rs[j];
#pragma unroll
        for (int t = 0; t < 4; ++t)
#pragma unroll
            for (int j = 0; j < 8; ++j) O[t][j] *= corr[j];

        // O += P * V  (V B-operand: lanes hold keys 8h..8h+7, 16+8h..23+8h (h = lane/16) from LDS)
#pragma unroll
        for (int dsub = 0; dsub < 4; ++dsub) {
            const int d = dsub * 16 + l15;
            const bf16* vp = &Vl[d][hiHalf ? 8 : 0];
            O[dsub] = wmma_bf16(pa, ld16(vp, vp + 16), O[dsub]);
        }
    }

    // epilogue: O /= l, store remerged heads [b, s, h*64+d]
    const int bb = bh / HEADS, h = bh % HEADS;
#pragma unroll
    for (int dsub = 0; dsub < 4; ++dsub) {
        const int d = h * DK + dsub * 16 + l15;
#pragma unroll
        for (int j = 0; j < 8; ++j) {
            const int s = q0 + j + (hiHalf ? 8 : 0);
            attn[(size_t)(bb * SDIM + s) * DM + d] = (bf16)(O[dsub][j] / lrun[j]);
        }
    }
}

// ---------------------------------------------------------------- launcher
extern "C" void kernel_launch(void* const* d_in, const int* in_sizes, int n_in,
                              void* d_out, int out_size, void* d_ws, size_t ws_size,
                              hipStream_t stream) {
    const float* q_in = (const float*)d_in[0];
    const float* k_in = (const float*)d_in[1];
    const float* v_in = (const float*)d_in[2];
    const float* wq = (const float*)d_in[3];  const float* bq = (const float*)d_in[4];
    const float* wk = (const float*)d_in[5];  const float* bk = (const float*)d_in[6];
    const float* wv = (const float*)d_in[7];  const float* bv = (const float*)d_in[8];
    const float* wo = (const float*)d_in[9];  const float* bo = (const float*)d_in[10];

    const size_t SZX = (size_t)MROWS * DM;        // activation elems
    const size_t SZW = (size_t)DM * DM;           // weight elems
    const size_t SZH = (size_t)BDIM * HEADS * SDIM * DK;

    char* ws = (char*)d_ws;
    bf16* XQ = (bf16*)ws;                 ws += SZX * 2;
    bf16* XK = (bf16*)ws;                 ws += SZX * 2;
    bf16* XV = (bf16*)ws;                 ws += SZX * 2;
    bf16* WQ = (bf16*)ws;                 ws += SZW * 2;
    bf16* WK = (bf16*)ws;                 ws += SZW * 2;
    bf16* WV = (bf16*)ws;                 ws += SZW * 2;
    bf16* WO = (bf16*)ws;                 ws += SZW * 2;
    bf16* QH = (bf16*)ws;                 ws += SZH * 2;
    bf16* KH = (bf16*)ws;                 ws += SZH * 2;
    bf16* VT = (bf16*)ws;                 ws += SZH * 2;
    bf16* AT = (bf16*)ws;                 ws += SZX * 2;

    // 1) precision conversion
    cvt_kernel<<<2048, 256, 0, stream>>>(q_in, XQ, (int)SZX);
    cvt_kernel<<<2048, 256, 0, stream>>>(k_in, XK, (int)SZX);
    cvt_kernel<<<2048, 256, 0, stream>>>(v_in, XV, (int)SZX);
    cvt_kernel<<<512, 256, 0, stream>>>(wq, WQ, (int)SZW);
    cvt_kernel<<<512, 256, 0, stream>>>(wk, WK, (int)SZW);
    cvt_kernel<<<512, 256, 0, stream>>>(wv, WV, (int)SZW);
    cvt_kernel<<<512, 256, 0, stream>>>(wo, WO, (int)SZW);

    // 2) QKV projections: (MROWS/16)*(DM/64) waves, 8 waves per block
    const int proj_blocks = (MROWS / 16) * (DM / 64) / 8;   // 768
    proj_kernel<0><<<proj_blocks, 256, 0, stream>>>(XQ, WQ, bq, QH);
    proj_kernel<0><<<proj_blocks, 256, 0, stream>>>(XK, WK, bk, KH);
    proj_kernel<2><<<proj_blocks, 256, 0, stream>>>(XV, WV, bv, VT);

    // 3) attention: 4 waves per block, 64 queries per block
    attn_kernel<<<BDIM * HEADS * (SDIM / 64), 128, 0, stream>>>(QH, KH, VT, AT);

    // 4) output projection -> fp32 d_out
    proj_kernel<3><<<proj_blocks, 256, 0, stream>>>(AT, WO, bo, (float*)d_out);

    (void)in_sizes; (void)n_in; (void)out_size; (void)ws_size;
}
